// ProGenAttention_82308753260980
// MI455X (gfx1250) — matrix operand order repaired and checked by reading, not compiled, not run on hardware
//
#include <hip/hip_runtime.h>

// ---------------------------------------------------------------------------
// ProGen attention layer for MI455X (gfx1250, wave32, WMMA).
// Pipeline: f32->bf16 converts, QKV GEMM (WMMA bf16 + async-LDS staging),
// RoPE+split, flash attention (WMMA bf16), output GEMM (WMMA bf16 -> f32).
// ---------------------------------------------------------------------------

typedef __attribute__((ext_vector_type(16))) __bf16 v16bf;
typedef __attribute__((ext_vector_type(8)))  __bf16 v8bf;
typedef __attribute__((ext_vector_type(8)))  float  v8f;

#define HIDDEN   4096
#define NHEADS   32
#define DHEAD    128
#define SLEN     2048
#define BATCH    2
#define TOKENS   (BATCH * SLEN)        // 4096
#define QKV_N    (3 * HIDDEN)          // 12288

#define KCHUNK    64                   // K depth staged per double-buffer step
#define LDS_PITCH 72                   // bf16 elements per staged row (64 data + 8 pad)

__device__ __forceinline__ v8f wmma_bf16(v16bf a, v16bf b, v8f c) {
    return __builtin_amdgcn_wmma_f32_16x16x32_bf16(
        /*neg_a=*/false, a, /*neg_b=*/false, b,
        /*c_mod=*/(short)0, c, /*reuse_a=*/false, /*reuse_b=*/false);
}

__device__ __forceinline__ v16bf combine8(v8bf lo, v8bf hi) {
    v16bf a;
#pragma unroll
    for (int i = 0; i < 8; ++i) { a[i] = lo[i]; a[i + 8] = hi[i]; }
    return a;
}

// A-fragment (16x32 bf16, M x K) from a contiguous row:
// lanes 0-15 : elems 0..15 = K 0..15
// lanes 16-31: elems 0..15 = K 16..31
__device__ __forceinline__ v16bf load_afrag(const __bf16* row, int kbase, int laneHi) {
    const __bf16* p = row + kbase + laneHi * 16;
    return combine8(*(const v8bf*)(p), *(const v8bf*)(p + 8));
}

// CDNA5 async copy: 16 bytes global -> LDS, tracked by ASYNCcnt.
__device__ __forceinline__ void async_copy_16(unsigned lds_off, const __bf16* g) {
    asm volatile("global_load_async_to_lds_b128 %0, %1, off"
                 :: "v"(lds_off), "v"(g) : "memory");
}
__device__ __forceinline__ void wait_async0() {
    asm volatile("s_wait_asynccnt 0" ::: "memory");
}

// ---------------------------------------------------------------------------
// Converts
// ---------------------------------------------------------------------------
__global__ void f32_to_bf16_kernel(const float* __restrict__ x,
                                   __bf16* __restrict__ y, size_t n) {
    size_t i = (size_t)blockIdx.x * blockDim.x + threadIdx.x;
    if (i < n) y[i] = (__bf16)x[i];
}

// Wt[n*K + k] = (bf16) W[k*N + n]   (coalesced read, strided write)
__global__ void transpose_f32_to_bf16_kernel(const float* __restrict__ W,
                                             __bf16* __restrict__ Wt,
                                             int K, int N) {
    size_t idx = (size_t)blockIdx.x * blockDim.x + threadIdx.x;
    if (idx >= (size_t)K * N) return;
    int k = (int)(idx / N);
    int n = (int)(idx % N);
    Wt[(size_t)n * K + k] = (__bf16)W[idx];
}

// ---------------------------------------------------------------------------
// GEMM: C[M,N] = A[M,K] * Bt[N,K]^T  (bf16 in, f32 accumulate, OT out)
// Block (8 waves) computes a 128x128 tile; K-chunks of 64 are staged into
// LDS with global_load_async_to_lds_b128 (double-buffered, ASYNCcnt +
// workgroup barriers; 16 WMMAs per barrier pair). Wave tile 32x64.
// ---------------------------------------------------------------------------
template <typename OT>
__global__ __launch_bounds__(256)
void gemm_wmma_kernel(const __bf16* __restrict__ A,
                      const __bf16* __restrict__ Bt,
                      OT* __restrict__ C,
                      int M, int N, int Kdim) {
    const int lane   = threadIdx.x & 31;
    const int wave   = threadIdx.x >> 5;
    const int ln     = lane & 15;
    const int laneHi = lane >> 4;
    const int wrow   = wave & 3;   // 4 waves along M (32 rows each)
    const int wcol   = wave >> 2;  // 2 waves along N (64 cols each)

    const int tilesN = N / 128;
    const int m0 = (blockIdx.x / tilesN) * 128;
    const int n0 = (blockIdx.x % tilesN) * 128;

    // 128 rows x (64 data + 8 pad) bf16, double-buffered, for A panel & B panel
    __shared__ __bf16 Asl[2][128 * LDS_PITCH];
    __shared__ __bf16 Bsl[2][128 * LDS_PITCH];

    // 256 threads stage 16KB of A + 16KB of B per chunk: 4x16B pieces each,
    // consecutive lanes copying consecutive 16B pieces (coalesced).
    auto stage = [&](int buf, int kk) {
#pragma unroll
        for (int p = 0; p < 4; ++p) {
            const int piece = (int)threadIdx.x + 256 * p;  // 0..1023
            const int row = piece >> 3;                    // 0..127
            const int q   = piece & 7;                     // 16B eighth of a 128B row
            const __bf16* ga = A  + (size_t)(m0 + row) * Kdim + kk + q * 8;
            const __bf16* gb = Bt + (size_t)(n0 + row) * Kdim + kk + q * 8;
            async_copy_16((unsigned)(size_t)&Asl[buf][row * LDS_PITCH + q * 8], ga);
            async_copy_16((unsigned)(size_t)&Bsl[buf][row * LDS_PITCH + q * 8], gb);
        }
    };

    v8f acc[2][4];
#pragma unroll
    for (int i = 0; i < 2; ++i)
#pragma unroll
        for (int j = 0; j < 4; ++j)
#pragma unroll
            for (int r = 0; r < 8; ++r) acc[i][j][r] = 0.0f;

    int buf = 0;
    stage(0, 0);
    for (int kk = 0; kk < Kdim; kk += KCHUNK) {
        wait_async0();       // own async copies into `buf` have landed
        __syncthreads();     // everyone's copies visible to all waves
        if (kk + KCHUNK < Kdim) stage(buf ^ 1, kk + KCHUNK);

        const __bf16* abase = &Asl[buf][0];
        const __bf16* bbase = &Bsl[buf][0];

#pragma unroll
        for (int kc = 0; kc < 2; ++kc) {       // two 32-deep sub-chunks
            // A fragments (rows of the wave's 32-row sub-panel)
            v16bf afr[2];
#pragma unroll
            for (int i = 0; i < 2; ++i) {
                const __bf16* arow = abase + (wrow * 32 + i * 16 + ln) * LDS_PITCH
                                           + kc * 32 + laneHi * 16;
                afr[i] = combine8(*(const v8bf*)(arow), *(const v8bf*)(arow + 8));
            }
#pragma unroll
            for (int j = 0; j < 4; ++j) {
                // B-fragment (32x16): lane holds col = lane%16, 16 contiguous K
                const __bf16* brow = bbase + (wcol * 64 + j * 16 + ln) * LDS_PITCH
                                           + kc * 32 + laneHi * 16;
                v16bf b = combine8(*(const v8bf*)(brow), *(const v8bf*)(brow + 8));
                acc[0][j] = wmma_bf16(afr[0], b, acc[0][j]);
                acc[1][j] = wmma_bf16(afr[1], b, acc[1][j]);
            }
        }
        __syncthreads();     // done reading; alternate buffer may be refilled
        buf ^= 1;
    }

    // C layout: VGPR r -> row r (lanes 0-15) / row r+8 (lanes 16-31); col = lane%16
#pragma unroll
    for (int i = 0; i < 2; ++i)
#pragma unroll
        for (int j = 0; j < 4; ++j)
#pragma unroll
            for (int r = 0; r < 8; ++r) {
                int row = m0 + wrow * 32 + i * 16 + r + 8 * laneHi;
                int col = n0 + wcol * 64 + j * 16 + ln;
                C[(size_t)row * N + col] = (OT)acc[i][j][r];
            }
}

// ---------------------------------------------------------------------------
// RoPE (NeoX, rotary_dim=64) + split into attention-friendly layouts:
//   Q [B,H,S,D], K [B,H,S,D], Vt [B,H,D,S]
// ---------------------------------------------------------------------------
__global__ void rope_split_kernel(const __bf16* __restrict__ qkv,
                                  const int* __restrict__ positions,
                                  __bf16* __restrict__ Q,
                                  __bf16* __restrict__ K,
                                  __bf16* __restrict__ Vt) {
    size_t idx = (size_t)blockIdx.x * blockDim.x + threadIdx.x;
    if (idx >= (size_t)TOKENS * QKV_N) return;
    int t = (int)(idx / QKV_N);
    int c = (int)(idx % QKV_N);
    int b = t >> 11, s = t & (SLEN - 1);
    int which = c / HIDDEN;
    int hd = c % HIDDEN;
    int h = hd >> 7, d = hd & 127;

    float x = (float)qkv[idx];
    if (which == 2) {
        Vt[(((size_t)(b * NHEADS + h)) * DHEAD + d) * SLEN + s] = (__bf16)x;
        return;
    }
    float outv = x;
    if (d < 64) {
        int i = d & 31;  // frequency index
        float inv = __powf(10000.0f, -(float)i / 32.0f);
        float ang = (float)positions[t] * inv;
        float sn, cs;
        __sincosf(ang, &sn, &cs);
        if (d < 32) {
            float x2 = (float)qkv[idx + 32];
            outv = x * cs - x2 * sn;
        } else {
            float x1 = (float)qkv[idx - 32];
            outv = x * cs + x1 * sn;
        }
    }
    size_t dst = (((size_t)(b * NHEADS + h)) * SLEN + s) * DHEAD + d;
    if (which == 0) Q[dst] = (__bf16)outv;
    else            K[dst] = (__bf16)outv;
}

// ---------------------------------------------------------------------------
// Flash attention: one wave per (b, h, 16-query tile).
// Streaming softmax kept in the WMMA C layout (row = acc index per half-wave),
// P reshaped C-layout -> A-fragment through a 1 KB per-wave LDS slab.
// ---------------------------------------------------------------------------
__global__ __launch_bounds__(128)
void flash_attn_kernel(const __bf16* __restrict__ Q,
                       const __bf16* __restrict__ K,
                       const __bf16* __restrict__ Vt,
                       __bf16* __restrict__ attn) {
    const int lane   = threadIdx.x & 31;
    const int wave   = threadIdx.x >> 5;
    const int ln     = lane & 15;
    const int laneHi = lane >> 4;

    const int job = blockIdx.x * 4 + wave;   // 8192 jobs total
    const int qt  = job & 127;               // query tile within sequence
    const int bh  = job >> 7;                // (b*32 + h)
    const int q0  = qt * 16;

    const __bf16* Qb = Q  + (size_t)bh * SLEN * DHEAD;
    const __bf16* Kb = K  + (size_t)bh * SLEN * DHEAD;
    const __bf16* Vb = Vt + (size_t)bh * DHEAD * SLEN;

    // Q tile (16x128) as 4 A-fragments, resident in registers.
    v16bf qf[4];
    {
        const __bf16* qrow = Qb + (size_t)(q0 + ln) * DHEAD;
#pragma unroll
        for (int kc = 0; kc < 4; ++kc) qf[kc] = load_afrag(qrow, kc * 32, laneHi);
    }

    v8f o[8];
#pragma unroll
    for (int nt = 0; nt < 8; ++nt)
#pragma unroll
        for (int r = 0; r < 8; ++r) o[nt][r] = 0.0f;

    float mrow[8], lrow[8];
#pragma unroll
    for (int r = 0; r < 8; ++r) { mrow[r] = -3.0e38f; lrow[r] = 0.0f; }

    __shared__ __bf16 plds[4][16 * 32];
    __bf16* pw = &plds[wave][0];

    const float scale = 0.08838834764831845f;  // 1/sqrt(128)
    const int qmax = q0 + 15;

    for (int kb = 0; kb <= qmax; kb += 32) {
        // ---- scores: two 16x16 tiles over this 32-key block ----
        v8f st[2];
#pragma unroll
        for (int tt = 0; tt < 2; ++tt) {
            v8f c;
#pragma unroll
            for (int r = 0; r < 8; ++r) c[r] = 0.0f;
            const int key = kb + tt * 16 + ln;
            const __bf16* krow = Kb + (size_t)key * DHEAD;
#pragma unroll
            for (int kc = 0; kc < 4; ++kc) {
                v16bf b = *(const v16bf*)(krow + kc * 32 + laneHi * 16);
                c = wmma_bf16(qf[kc], b, c);
            }
            st[tt] = c;
        }

        // ---- scale + causal mask + per-row block max ----
        float bmax[8];
#pragma unroll
        for (int r = 0; r < 8; ++r) {
            const int qrow = q0 + r + 8 * laneHi;
            float m = -3.0e38f;
#pragma unroll
            for (int tt = 0; tt < 2; ++tt) {
                const int key = kb + tt * 16 + ln;
                float s = st[tt][r] * scale;
                if (key > qrow) s = -3.0e38f;
                st[tt][r] = s;
                m = fmaxf(m, s);
            }
            bmax[r] = m;
        }

        // ---- running softmax update (row lives in a 16-lane half) ----
#pragma unroll
        for (int r = 0; r < 8; ++r) {
            float m = bmax[r];
#pragma unroll
            for (int off = 1; off < 16; off <<= 1)
                m = fmaxf(m, __shfl_xor(m, off, 32));
            const float mn = fmaxf(mrow[r], m);
            const float corr = __expf(mrow[r] - mn);
#pragma unroll
            for (int nt = 0; nt < 8; ++nt) o[nt][r] *= corr;
            lrow[r] *= corr;
            mrow[r] = mn;
        }

        // ---- probabilities -> LDS (C layout scatter), row sums ----
#pragma unroll
        for (int r = 0; r < 8; ++r) {
            const int row = r + 8 * laneHi;
            float rsum = 0.0f;
#pragma unroll
            for (int tt = 0; tt < 2; ++tt) {
                float p = __expf(st[tt][r] - mrow[r]);
                rsum += p;
                pw[row * 32 + tt * 16 + ln] = (__bf16)p;
            }
#pragma unroll
            for (int off = 1; off < 16; off <<= 1)
                rsum += __shfl_xor(rsum, off, 32);
            lrow[r] += rsum;
        }

        // ---- re-read P as A-fragment (16x32) ----
        v16bf pf;
        {
            const __bf16* prow = pw + ln * 32 + laneHi * 16;
#pragma unroll
            for (int i = 0; i < 8; ++i) { pf[i] = prow[i]; pf[i + 8] = prow[i + 8]; }
        }

        // ---- O += P * V  (8 d-tiles of 16) ----
#pragma unroll
        for (int nt = 0; nt < 8; ++nt) {
            const int d = nt * 16 + ln;
            const __bf16* vrow = Vb + (size_t)d * SLEN + kb + laneHi * 16;
            v16bf b = *(const v16bf*)(vrow);
            o[nt] = wmma_bf16(pf, b, o[nt]);
        }
    }

    // ---- normalize + store attn in [token, h*128+d] for the output GEMM ----
#pragma unroll
    for (int r = 0; r < 8; ++r) {
        const float inv = 1.0f / lrow[r];
#pragma unroll
        for (int nt = 0; nt < 8; ++nt) o[nt][r] *= inv;
    }
    const int b = bh >> 5, h = bh & 31;
#pragma unroll
    for (int nt = 0; nt < 8; ++nt) {
        const int d = nt * 16 + ln;
#pragma unroll
        for (int r = 0; r < 8; ++r) {
            const int s = q0 + r + 8 * laneHi;
            attn[((size_t)(b * SLEN + s)) * HIDDEN + h * DHEAD + d] = (__bf16)o[nt][r];
        }
    }
}

// ---------------------------------------------------------------------------
// Launch
// ---------------------------------------------------------------------------
extern "C" void kernel_launch(void* const* d_in, const int* in_sizes, int n_in,
                              void* d_out, int out_size, void* d_ws, size_t ws_size,
                              hipStream_t stream) {
    const float* hidden    = (const float*)d_in[0];
    const int*   positions = (const int*)d_in[1];
    const float* Wqkv      = (const float*)d_in[2];
    const float* Wout      = (const float*)d_in[3];
    float*       out       = (float*)d_out;

    char* ws = (char*)d_ws;
    __bf16* hid_bf = (__bf16*)(ws);                       //  33,554,432 B
    __bf16* WqkvT  = (__bf16*)(ws + 33554432ull);         // 100,663,296 B
    __bf16* WoutT  = (__bf16*)(ws + 134217728ull);        //  33,554,432 B
    __bf16* qkv    = (__bf16*)(ws + 167772160ull);        // 100,663,296 B
    __bf16* Qm     = (__bf16*)(ws + 268435456ull);        //  33,554,432 B
    __bf16* Km     = (__bf16*)(ws + 301989888ull);        //  33,554,432 B
    __bf16* Vt     = (__bf16*)(ws + 335544320ull);        //  33,554,432 B
    __bf16* attn   = (__bf16*)(ws + 369098752ull);        //  33,554,432 B

    // 1) converts
    f32_to_bf16_kernel<<<65536, 256, 0, stream>>>(hidden, hid_bf,
                                                  (size_t)TOKENS * HIDDEN);
    transpose_f32_to_bf16_kernel<<<196608, 256, 0, stream>>>(Wqkv, WqkvT,
                                                             HIDDEN, QKV_N);
    transpose_f32_to_bf16_kernel<<<65536, 256, 0, stream>>>(Wout, WoutT,
                                                            HIDDEN, HIDDEN);
    // 2) QKV projection: (4096 x 4096) * (4096 x 12288), 128x128 block tiles
    gemm_wmma_kernel<__bf16><<<3072, 256, 0, stream>>>(hid_bf, WqkvT, qkv,
                                                       TOKENS, QKV_N, HIDDEN);
    // 3) RoPE + split into Q/K/Vt
    rope_split_kernel<<<196608, 256, 0, stream>>>(qkv, positions, Qm, Km, Vt);
    // 4) causal flash attention (one wave per 16-query tile)
    flash_attn_kernel<<<2048, 128, 0, stream>>>(Qm, Km, Vt, attn);
    // 5) output projection: (4096 x 4096) * (4096 x 4096) -> f32
    gemm_wmma_kernel<float><<<1024, 256, 0, stream>>>(attn, WoutT, out,
                                                      TOKENS, HIDDEN, HIDDEN);
}
